// KGEncoder_75806172775027
// MI455X (gfx1250) — compile-verified, matrix operand order checked by reading, not run on hardware
//
#include <hip/hip_runtime.h>
#include <hip/hip_bf16.h>
#include <math.h>

typedef __attribute__((ext_vector_type(16))) __bf16 v16bf;
typedef __attribute__((ext_vector_type(8)))  __bf16 v8bf;
typedef __attribute__((ext_vector_type(8)))  float  v8f;

#define NN 32768
#define EE 262144
#define DD 128
#define HH 4
#define LL 2
#define BB 256
#define NRr 500
#define NRp 512
#define DD2 (128*128)

// ---------------------------------------------------------------------------
// Fragment load: ISA 16-bit A/B layout. Per lane the 16 elements map to
// k-offsets {0..7} and {16..23} relative to (base + 8*half) -> two contiguous
// 16-byte runs -> two global_load_b128.
// ---------------------------------------------------------------------------
__device__ inline v16bf load_frag16(const __bf16* __restrict__ p) {
  v8bf lo = *(const v8bf*)(p);        // k-offsets 0..7
  v8bf hi = *(const v8bf*)(p + 16);   // k-offsets 16..23
  return __builtin_shufflevector(lo, hi, 0, 1, 2, 3, 4, 5, 6, 7,
                                         8, 9, 10, 11, 12, 13, 14, 15);
}

// ---------------------------------------------------------------------------
// WMMA GEMM: C[M,128] = A[M,128](bf16,row-major) x B (given TRANSPOSED:
// Bt[128 n][128 k] row-major). One wave per 16x16 C tile, 4 x wmma bf16.
// ---------------------------------------------------------------------------
__global__ __launch_bounds__(256)
void wmma_gemm_128t(const __bf16* __restrict__ A, const __bf16* __restrict__ Bt,
                    float* __restrict__ C, int M) {
  int wave = (blockIdx.x * blockDim.x + threadIdx.x) >> 5;
  int lane = threadIdx.x & 31;
  int mt = wave >> 3;          // 8 N-tiles (128/16)
  int nt = wave & 7;
  if (mt * 16 >= M) return;
  int half = lane >> 4;
  int l15 = lane & 15;
  const __bf16* Ap = A  + (size_t)(mt * 16 + l15) * 128 + 8 * half;  // row m
  const __bf16* Bp = Bt + (size_t)(nt * 16 + l15) * 128 + 8 * half;  // col n
  v8f acc = {};
#pragma unroll
  for (int kk = 0; kk < 128; kk += 32) {
    v16bf a = load_frag16(Ap + kk);
    v16bf b = load_frag16(Bp + kk);
    acc = __builtin_amdgcn_wmma_f32_16x16x32_bf16(false, a, false, b,
                                                  (short)0, acc, false, false);
  }
  float* Crow = C + (size_t)(mt * 16 + 8 * half) * 128 + nt * 16 + l15;
#pragma unroll
  for (int i = 0; i < 8; ++i) Crow[(size_t)i * 128] = acc[i];
}

// ---------------------------------------------------------------------------
// small helpers
// ---------------------------------------------------------------------------
__device__ inline float gelu_tanh(float x) {
  const float c = 0.7978845608028654f; // sqrt(2/pi)
  float t = tanhf(c * (x + 0.044715f * x * x * x));
  return 0.5f * x * (1.0f + t);
}

__device__ inline void atomicMaxF(float* addr, float val) {
  unsigned* ua = (unsigned*)addr;
  unsigned old = *ua;
  for (;;) {
    float f = __uint_as_float(old);
    if (f >= val) break;
    unsigned assumed = old;
    old = atomicCAS(ua, assumed, __float_as_uint(val));
    if (old == assumed) break;
  }
}

// weights: f32 [l][k][n] -> bf16 TRANSPOSED [l][n][k] (coalesced writes)
__global__ void cvt_w_bf16_t(const float* __restrict__ W, __bf16* __restrict__ Wt) {
  int i = blockIdx.x * blockDim.x + threadIdx.x;   // LL*DD2
  if (i >= LL * DD2) return;
  int l = i / DD2, r = i - l * DD2;
  int n = r >> 7, k = r & 127;
  Wt[i] = (__bf16)W[(size_t)l * DD2 + (size_t)k * DD + n];
}

__global__ void cvt_f32_bf16(const float* __restrict__ in, __bf16* __restrict__ out, int n) {
  int i = blockIdx.x * blockDim.x + threadIdx.x;
  if (i < n) out[i] = (__bf16)in[i];
}

__global__ void cvt_rel_pad(const float* __restrict__ rel, __bf16* __restrict__ relb) {
  int i = blockIdx.x * blockDim.x + threadIdx.x;   // NRp*DD
  if (i >= NRp * DD) return;
  int r = i >> 7;
  relb[i] = (__bf16)((r < NRr) ? rel[i] : 0.0f);
}

__global__ void gather_nodes(const float* __restrict__ ent, const int* __restrict__ x,
                             float* __restrict__ h, __bf16* __restrict__ hb) {
  int i = blockIdx.x * blockDim.x + threadIdx.x;   // NN*DD
  if (i >= NN * DD) return;
  int n = i >> 7, d = i & 127;
  float v = ent[(size_t)x[n] * DD + d];
  h[i] = v;
  hb[i] = (__bf16)v;
}

__global__ void init_layer(float* __restrict__ m, float* __restrict__ den,
                           float* __restrict__ msg) {
  int i = blockIdx.x * blockDim.x + threadIdx.x;   // NN*DD
  if (i < NN * DD) msg[i] = 0.0f;
  if (i < NN * HH) { m[i] = -INFINITY; den[i] = 0.0f; }
}

// ---------------------------------------------------------------------------
// edge stage (memory/atomic bound)
// ---------------------------------------------------------------------------
__global__ void edge_score(const int* __restrict__ src, const int* __restrict__ dst,
                           const int* __restrict__ ea,
                           const float* __restrict__ q, const float* __restrict__ k,
                           const float* __restrict__ ek, const float* __restrict__ prior,
                           float* __restrict__ score, float* __restrict__ m) {
  int idx = blockIdx.x * blockDim.x + threadIdx.x;   // EE*HH
  if (idx >= EE * HH) return;
  int e = idx >> 2, hh = idx & 3;
  int sN = src[e], dN = dst[e], r = ea[e];
  const float4* qp = (const float4*)(q + (size_t)dN * DD + hh * 32);
  const float4* kp = (const float4*)(k + (size_t)sN * DD + hh * 32);
  const float4* ep = (const float4*)(ek + (size_t)r * DD + hh * 32);
  float acc = 0.0f;
#pragma unroll
  for (int j = 0; j < 8; ++j) {
    float4 qv = qp[j], kv = kp[j], ev = ep[j];
    acc += qv.x * (kv.x + ev.x) + qv.y * (kv.y + ev.y)
         + qv.z * (kv.z + ev.z) + qv.w * (kv.w + ev.w);
  }
  float s = acc * 0.17677669529663687f * prior[r];  // 1/sqrt(32) * rel_prior
  score[idx] = s;
  atomicMaxF(&m[dN * HH + hh], s);
}

__global__ void edge_exp(const int* __restrict__ dst, float* __restrict__ score,
                         const float* __restrict__ m, float* __restrict__ den) {
  int idx = blockIdx.x * blockDim.x + threadIdx.x;   // EE*HH
  if (idx >= EE * HH) return;
  int e = idx >> 2, hh = idx & 3;
  int dN = dst[e];
  float mv = m[dN * HH + hh];
  if (mv == -INFINITY) mv = 0.0f;                    // isfinite() guard from ref
  float ex = expf(score[idx] - mv);
  score[idx] = ex;                                   // overwrite score with exp
  atomicAdd(&den[dN * HH + hh], ex);
}

__global__ void edge_scatter(const int* __restrict__ src, const int* __restrict__ dst,
                             const int* __restrict__ ea,
                             const float* __restrict__ score, const float* __restrict__ den,
                             const float* __restrict__ v, const float* __restrict__ ev,
                             float* __restrict__ msg) {
  int idx = blockIdx.x * blockDim.x + threadIdx.x;   // EE * (DD/4)
  if (idx >= EE * 32) return;
  int e = idx >> 5, c = idx & 31;
  int d0 = c << 2, hh = d0 >> 5;
  int sN = src[e], dN = dst[e], r = ea[e];
  float alpha = score[e * HH + hh] / (den[dN * HH + hh] + 1e-16f);
  float4 vv = *(const float4*)(v + (size_t)sN * DD + d0);
  float4 eev = *(const float4*)(ev + (size_t)r * DD + d0);
  float* mp = msg + (size_t)dN * DD + d0;
  atomicAdd(mp + 0, alpha * (vv.x + eev.x));
  atomicAdd(mp + 1, alpha * (vv.y + eev.y));
  atomicAdd(mp + 2, alpha * (vv.z + eev.z));
  atomicAdd(mp + 3, alpha * (vv.w + eev.w));
}

__global__ void residual_gelu(const float* __restrict__ mo, float* __restrict__ h,
                              __bf16* __restrict__ hb) {
  int i = blockIdx.x * blockDim.x + threadIdx.x;     // NN*DD
  if (i >= NN * DD) return;
  float g = gelu_tanh(h[i] + mo[i]);
  h[i] = g;
  hb[i] = (__bf16)g;
}

// ---------------------------------------------------------------------------
// readout + loss
// ---------------------------------------------------------------------------
__global__ void gather_out(const float* __restrict__ h, const int* __restrict__ y,
                           const int* __restrict__ s, float* __restrict__ emb) {
  int b = blockIdx.x, d = threadIdx.x;               // BB blocks x 128 threads
  int off = 0;
  for (int i = 0; i < b; ++i) off += s[i];           // exclusive cumsum
  int idx = off + y[b];
  emb[(size_t)b * DD + d] = h[(size_t)idx * DD + d];
}

__global__ void pos_neg_kernel(const float* __restrict__ eh, const float* __restrict__ et,
                               const float* __restrict__ en, const float* __restrict__ rel,
                               const int* __restrict__ sample,
                               float* __restrict__ pos, float* __restrict__ neg) {
  int b = blockIdx.x, d = threadIdx.x;               // BB blocks x 128 threads
  int r = sample[b * 3 + 1];
  float tv = et[(size_t)b * DD + d];
  float dp = tv - (eh[(size_t)b * DD + d] + rel[(size_t)r * DD + d]) + 1e-8f;
  float dn = tv - en[(size_t)b * DD + d] + 1e-8f;
  __shared__ float sp[DD], sn[DD];
  sp[d] = dp * dp; sn[d] = dn * dn;
  __syncthreads();
  for (int st = 64; st > 0; st >>= 1) {
    if (d < st) { sp[d] += sp[d + st]; sn[d] += sn[d + st]; }
    __syncthreads();
  }
  if (d == 0) { pos[b] = sqrtf(sp[0]); neg[b] = sqrtf(sn[0]); }
}

__global__ void loss_kernel(const float* __restrict__ pos, const float* __restrict__ neg,
                            float* __restrict__ out) {
  int i = threadIdx.x;                                // 256 threads, 1 block
  float p = pos[i], acc = 0.0f;
  for (int j = 0; j < BB; ++j) acc += fmaxf(p - neg[j] + 1.0f, 0.0f);
  __shared__ float sh[BB];
  sh[i] = acc;
  __syncthreads();
  for (int st = 128; st > 0; st >>= 1) {
    if (i < st) sh[i] += sh[i + st];
    __syncthreads();
  }
  if (i == 0) out[0] = sh[0] / (float)(BB * BB);
}

// ---------------------------------------------------------------------------
// host
// ---------------------------------------------------------------------------
extern "C" void kernel_launch(void* const* d_in, const int* in_sizes, int n_in,
                              void* d_out, int out_size, void* d_ws, size_t ws_size,
                              hipStream_t stream) {
  const float* ent_emb   = (const float*)d_in[0];
  const float* rel_emb   = (const float*)d_in[1];
  const float* rel_prior = (const float*)d_in[2];
  const float* W[6] = { (const float*)d_in[3], (const float*)d_in[4],
                        (const float*)d_in[5], (const float*)d_in[6],
                        (const float*)d_in[7], (const float*)d_in[8] };
  const int* sample = (const int*)d_in[24];

  // ---- workspace carve ----
  char* p = (char*)d_ws;
  auto carve = [&](size_t bytes) -> void* {
    void* r = (void*)p;
    p += (bytes + 255) & ~(size_t)255;
    return r;
  };
  __bf16* Wt[6];  // transposed bf16 weights (B operands)
  for (int w = 0; w < 6; ++w) Wt[w] = (__bf16*)carve((size_t)LL * DD2 * sizeof(__bf16));
  __bf16* relb  = (__bf16*)carve((size_t)NRp * DD * sizeof(__bf16));
  float*  ekbuf = (float*)carve((size_t)LL * NRp * DD * sizeof(float));
  float*  evbuf = (float*)carve((size_t)LL * NRp * DD * sizeof(float));
  float*  h     = (float*)carve((size_t)NN * DD * sizeof(float));
  __bf16* hb    = (__bf16*)carve((size_t)NN * DD * sizeof(__bf16));
  float*  q     = (float*)carve((size_t)NN * DD * sizeof(float));
  float*  k     = (float*)carve((size_t)NN * DD * sizeof(float));
  float*  v     = (float*)carve((size_t)NN * DD * sizeof(float));
  float*  msg   = (float*)carve((size_t)NN * DD * sizeof(float));
  __bf16* msgb  = (__bf16*)carve((size_t)NN * DD * sizeof(__bf16));
  float*  score = (float*)carve((size_t)EE * HH * sizeof(float));
  float*  mbuf  = (float*)carve((size_t)NN * HH * sizeof(float));
  float*  den   = (float*)carve((size_t)NN * HH * sizeof(float));
  float*  emb3  = (float*)carve((size_t)3 * BB * DD * sizeof(float));
  float*  posb  = (float*)carve((size_t)BB * sizeof(float));
  float*  negb  = (float*)carve((size_t)BB * sizeof(float));

  const int T = 256;
  const int gND = (NN * DD + T - 1) / T;
  const int gEH = (EE * HH + T - 1) / T;
  const int gE4 = (EE * 32 + T - 1) / T;

  // ---- one-time: weights -> bf16 transposed, relation projections ----
  for (int w = 0; w < 6; ++w)
    cvt_w_bf16_t<<<(LL * DD2 + T - 1) / T, T, 0, stream>>>(W[w], Wt[w]);
  cvt_rel_pad<<<(NRp * DD + T - 1) / T, T, 0, stream>>>(rel_emb, relb);
  for (int l = 0; l < LL; ++l) {
    wmma_gemm_128t<<<NRp / 16, T, 0, stream>>>(relb, Wt[4] + (size_t)l * DD2,
                                               ekbuf + (size_t)l * NRp * DD, NRp);
    wmma_gemm_128t<<<NRp / 16, T, 0, stream>>>(relb, Wt[5] + (size_t)l * DD2,
                                               evbuf + (size_t)l * NRp * DD, NRp);
  }

  // ---- three graphs: h, t, n ----
  for (int g = 0; g < 3; ++g) {
    const int* x  = (const int*)d_in[9 + 5 * g];
    const int* ei = (const int*)d_in[10 + 5 * g];
    const int* ea = (const int*)d_in[11 + 5 * g];
    const int* y  = (const int*)d_in[12 + 5 * g];
    const int* s  = (const int*)d_in[13 + 5 * g];
    const int* srcp = ei;
    const int* dstp = ei + EE;

    gather_nodes<<<gND, T, 0, stream>>>(ent_emb, x, h, hb);

    for (int l = 0; l < LL; ++l) {
      const __bf16* wq = Wt[0] + (size_t)l * DD2;
      const __bf16* wk = Wt[1] + (size_t)l * DD2;
      const __bf16* wv = Wt[2] + (size_t)l * DD2;
      const __bf16* wo = Wt[3] + (size_t)l * DD2;
      const float* ekl = ekbuf + (size_t)l * NRp * DD;
      const float* evl = evbuf + (size_t)l * NRp * DD;

      wmma_gemm_128t<<<NN / 16, T, 0, stream>>>(hb, wq, q, NN);
      wmma_gemm_128t<<<NN / 16, T, 0, stream>>>(hb, wk, k, NN);
      wmma_gemm_128t<<<NN / 16, T, 0, stream>>>(hb, wv, v, NN);

      init_layer<<<gND, T, 0, stream>>>(mbuf, den, msg);
      edge_score<<<gEH, T, 0, stream>>>(srcp, dstp, ea, q, k, ekl, rel_prior, score, mbuf);
      edge_exp<<<gEH, T, 0, stream>>>(dstp, score, mbuf, den);
      edge_scatter<<<gE4, T, 0, stream>>>(srcp, dstp, ea, score, den, v, evl, msg);

      cvt_f32_bf16<<<gND, T, 0, stream>>>(msg, msgb, NN * DD);
      wmma_gemm_128t<<<NN / 16, T, 0, stream>>>(msgb, wo, q, NN);  // reuse q as msg@Wo
      residual_gelu<<<gND, T, 0, stream>>>(q, h, hb);
    }
    gather_out<<<BB, DD, 0, stream>>>(h, y, s, emb3 + (size_t)g * BB * DD);
  }

  // ---- TransE margin-ranking loss ----
  pos_neg_kernel<<<BB, DD, 0, stream>>>(emb3, emb3 + (size_t)BB * DD,
                                        emb3 + (size_t)2 * BB * DD,
                                        rel_emb, sample, posb, negb);
  loss_kernel<<<1, BB, 0, stream>>>(posb, negb, (float*)d_out);
}
